// Decoder_10093173146074
// MI455X (gfx1250) — matrix operand order repaired and checked by reading, not compiled, not run on hardware
//
#include <hip/hip_runtime.h>

// LSTM decoder: B=256, T=1024, F=64, H=128, gates G=4H=512, fused K = F+H = 192.
// Kernel 1 (prep): convert [W_ih | W_hh] -> bf16 Wg[512][192], W_out -> bf16, bias sum.
// Kernel 2 (scan): 16 workgroups x 256 threads; each WG owns 16 batch rows, scans T
//   steps with LDS-resident bf16 weights, wmma_f32_16x16x32_bf16 gate GEMM (K=192),
//   register-resident cell state, fused bf16 output projection straight to d_out.

typedef __bf16 bf16_t;
typedef __attribute__((ext_vector_type(16))) __bf16 v16bf;
typedef __attribute__((ext_vector_type(8)))  __bf16 v8bf;
typedef __attribute__((ext_vector_type(8)))  float  v8f;

#define BB 256
#define TT 1024
#define FF 64
#define HH 128
#define GG 512          // 4*H
#define KK 192          // F + H
#define WS 200          // padded LDS row stride (bf16 elems) for weights / A staging
#define GS 520          // padded LDS row stride (f32) for gate exchange

__device__ __forceinline__ v8f wmma_bf16(v16bf a, v16bf b, v8f c) {
  // (neg_a, A, neg_b, B, c_mod, C, reuse_a, reuse_b)
  return __builtin_amdgcn_wmma_f32_16x16x32_bf16(false, a, false, b, (short)0, c, false, false);
}

// A fragment (16x32, M x K): lane m<16 holds row m, K=kb..kb+15 contiguous,
// kb = (lane>=16) ? 16 : 0 (matches load_B's K order).
__device__ __forceinline__ v16bf load_A(const bf16_t* base, int stride, int lane) {
  const int m  = lane & 15;
  const int kb = (lane >> 4) << 4;
  const bf16_t* p = base + m * stride + kb;
  v16bf a;
#pragma unroll
  for (int i = 0; i < 16; ++i) a[i] = p[i];
  return a;
}

// B fragment (32x16, K x N) with B[k][n] = W[n][k]: lane n<16 holds col n,
// K=0..15 (16 contiguous bf16 of W row n); lanes 16-31 hold K=16..31.
__device__ __forceinline__ v16bf load_B(const bf16_t* base, int stride, int lane) {
  const int n  = lane & 15;
  const int kb = (lane >> 4) << 4;
  const bf16_t* p = base + n * stride + kb;
  v16bf b;
#pragma unroll
  for (int i = 0; i < 16; ++i) b[i] = p[i];
  return b;
}

__global__ void prep_kernel(const float* __restrict__ W_ih, const float* __restrict__ W_hh,
                            const float* __restrict__ b_ih, const float* __restrict__ b_hh,
                            const float* __restrict__ W_out,
                            bf16_t* __restrict__ Wg, bf16_t* __restrict__ Wo,
                            float* __restrict__ bias) {
  const int tid = blockIdx.x * blockDim.x + threadIdx.x;
  const int nth = gridDim.x * blockDim.x;
  for (int i = tid; i < GG * KK; i += nth) {
    const int g = i / KK, k = i % KK;
    const float v = (k < FF) ? W_ih[g * FF + k] : W_hh[g * HH + (k - FF)];
    Wg[i] = (bf16_t)v;
  }
  for (int i = tid; i < FF * HH; i += nth) Wo[i] = (bf16_t)W_out[i];
  for (int i = tid; i < GG; i += nth) bias[i] = b_ih[i] + b_hh[i];
}

__launch_bounds__(256, 1)
__global__ void lstm_scan_kernel(const float* __restrict__ x,
                                 const bf16_t* __restrict__ Wg,
                                 const bf16_t* __restrict__ Wo,
                                 const float* __restrict__ bias,
                                 float* __restrict__ out) {
  __shared__ bf16_t sWg[GG * WS];    // 200 KB: fused weights, row-major [512][192]+pad
  __shared__ bf16_t sA[16 * WS];     // A staging: cols 0-63 = x_t (bf16), 64-191 = h
  __shared__ float  sGate[16 * GS];  // gate exchange [16][512]+pad
  __shared__ float  sBias[GG];

  const int tid  = threadIdx.x;
  const int lane = tid & 31;
  const int wave = tid >> 5;
  const int b0   = blockIdx.x * 16;

  // ---- one-time: weights + bias into LDS (16B vector copies) ----
  for (int i = tid; i < GG * (KK / 8); i += 256) {
    const int g = i / (KK / 8);
    const int k = (i % (KK / 8)) * 8;
    *(v8bf*)&sWg[g * WS + k] = *(const v8bf*)&Wg[g * KK + k];
  }
  for (int i = tid; i < GG; i += 256) sBias[i] = bias[i];

  // ---- W_out B-fragments register-resident (waves 0-3 only; N-slice = wave*16) ----
  v16bf woB[4];
  if (wave < 4) {
#pragma unroll
    for (int kk = 0; kk < 4; ++kk)
      woB[kk] = load_B(Wo + (wave * 16) * HH + kk * 32, HH, lane);
  }

  // ---- init: h=0 staged, c=0 in regs, stage x_0 ----
  for (int i = tid; i < 16 * HH; i += 256) {
    const int r = i / HH, c = i % HH;
    sA[r * WS + FF + c] = (bf16_t)0.0f;
  }
  for (int i = tid; i < 16 * FF; i += 256) {
    const int r = i / FF, c = i % FF;
    sA[r * WS + c] = (bf16_t)x[(size_t)(b0 + r) * TT * FF + c];
  }
  float c_reg[8];
#pragma unroll
  for (int i = 0; i < 8; ++i) c_reg[i] = 0.0f;
  __syncthreads();

  const int n0 = wave * 64;  // this wave's 64-wide gate column slice

  for (int t = 0; t < TT; ++t) {
    // ---- phase 1: gates[16,512] += [x_t|h] (16x192) @ Wg^T, bf16 WMMA ----
    v8f acc[4];
#pragma unroll
    for (int nn = 0; nn < 4; ++nn)
      acc[nn] = (v8f){0.f, 0.f, 0.f, 0.f, 0.f, 0.f, 0.f, 0.f};
#pragma unroll
    for (int kk = 0; kk < 6; ++kk) {
      const v16bf a = load_A(sA + kk * 32, WS, lane);
#pragma unroll
      for (int nn = 0; nn < 4; ++nn) {
        const v16bf b = load_B(sWg + (size_t)(n0 + nn * 16) * WS + kk * 32, WS, lane);
        acc[nn] = wmma_bf16(a, b, acc[nn]);
      }
    }
    // ---- phase 2: C frags -> LDS gate buffer (C layout: lane=col, vgpr r = row) ----
    {
      const int n_lo  = lane & 15;
      const int rbase = (lane >> 4) * 8;
#pragma unroll
      for (int nn = 0; nn < 4; ++nn) {
        const int n = n0 + nn * 16 + n_lo;
#pragma unroll
        for (int r = 0; r < 8; ++r) sGate[(rbase + r) * GS + n] = acc[nn][r];
      }
    }
    __syncthreads();

    // ---- phase 3: elementwise LSTM cell; wave owns hidden cols [16w,16w+16) ----
    {
      const int j  = wave * 16 + (lane & 15);
      const int r0 = (lane >> 4) * 8;
#pragma unroll
      for (int r = 0; r < 8; ++r) {
        const int row = r0 + r;
        float gi = sGate[row * GS + j]            + sBias[j];
        float gf = sGate[row * GS + HH + j]       + sBias[HH + j];
        float gg = sGate[row * GS + 2 * HH + j]   + sBias[2 * HH + j];
        float go = sGate[row * GS + 3 * HH + j]   + sBias[3 * HH + j];
        gi = 1.0f / (1.0f + __expf(-gi));
        gf = 1.0f / (1.0f + __expf(-gf));
        go = 1.0f / (1.0f + __expf(-go));
        gg = tanhf(gg);
        const float c = gf * c_reg[r] + gi * gg;
        c_reg[r] = c;
        const float h = go * tanhf(c);
        sA[row * WS + FF + j] = (bf16_t)h;  // restage h (bf16) for next step's A
      }
    }
    __syncthreads();

    // ---- phase 4: waves 0-3 -> out_t = h_t @ W_out^T (16x64); waves 4-7 -> stage x_{t+1}
    if (wave < 4) {
      v8f oacc = (v8f){0.f, 0.f, 0.f, 0.f, 0.f, 0.f, 0.f, 0.f};
#pragma unroll
      for (int kk = 0; kk < 4; ++kk) {
        const v16bf a = load_A(sA + FF + kk * 32, WS, lane);
        oacc = wmma_bf16(a, woB[kk], oacc);
      }
      const int n     = wave * 16 + (lane & 15);
      const int rbase = (lane >> 4) * 8;
#pragma unroll
      for (int r = 0; r < 8; ++r)
        out[(size_t)(b0 + rbase + r) * TT * FF + (size_t)t * FF + n] = oacc[r];
    } else if (t + 1 < TT) {
      const int id  = (wave - 4) * 32 + lane;  // 0..127
      const int row = id >> 3;
      const int c0  = (id & 7) * 8;
      const float* xp = x + (size_t)(b0 + row) * TT * FF + (size_t)(t + 1) * FF + c0;
#pragma unroll
      for (int i = 0; i < 8; ++i) sA[row * WS + c0 + i] = (bf16_t)xp[i];
    }
    __syncthreads();
  }
}

extern "C" void kernel_launch(void* const* d_in, const int* in_sizes, int n_in,
                              void* d_out, int out_size, void* d_ws, size_t ws_size,
                              hipStream_t stream) {
  const float* x     = (const float*)d_in[0];
  const float* W_ih  = (const float*)d_in[1];
  const float* W_hh  = (const float*)d_in[2];
  const float* b_ih  = (const float*)d_in[3];
  const float* b_hh  = (const float*)d_in[4];
  const float* W_out = (const float*)d_in[5];
  float* out = (float*)d_out;

  // workspace: Wg bf16 [512][192] | Wo bf16 [64][128] | bias f32 [512]  (~215 KB)
  bf16_t* Wg = (bf16_t*)d_ws;
  bf16_t* Wo = Wg + GG * KK;
  float*  bias = (float*)(Wo + FF * HH);

  prep_kernel<<<64, 256, 0, stream>>>(W_ih, W_hh, b_ih, b_hh, W_out, Wg, Wo, bias);
  lstm_scan_kernel<<<BB / 16, 256, 0, stream>>>(x, Wg, Wo, bias, out);
}
